// CCALoss_2997887172781
// MI455X (gfx1250) — 8-bit matrix operand order repaired and checked by reading, compiled, not run on hardware
//
#include <hip/hip_runtime.h>
#include <hip/hip_bf16.h>

typedef int v8i __attribute__((ext_vector_type(8)));

#define NEG_BIG (-3.402823466e38f)

__global__ void zero_acc_kernel(float* acc) {
    if (threadIdx.x < 4) acc[threadIdx.x] = 0.0f;
}

__global__ __launch_bounds__(256) void pack_bce_kernel(
    const int* __restrict__ mc, const float* __restrict__ clog,
    signed char* __restrict__ c8, int* __restrict__ rowsum,
    float* __restrict__ acc)
{
    __shared__ int   s_i[256];
    __shared__ float s_b[256];
    __shared__ float s_m[256];

    const int row = blockIdx.x;
    const int tid = threadIdx.x;
    const int idx = row * 256 + tid;

    const int v   = mc[idx];
    const int bin = (v == 1) ? 1 : 0;
    c8[idx] = (signed char)bin;

    const float mask = (v != -1) ? 1.0f : 0.0f;
    const float x = clog[idx];
    const float t = (float)bin;
    const float bce = fmaxf(x, 0.0f) - x * t + log1pf(expf(-fabsf(x)));

    s_i[tid] = bin;
    s_b[tid] = bce * mask;
    s_m[tid] = mask;
    __syncthreads();
    for (int off = 128; off > 0; off >>= 1) {
        if (tid < off) {
            s_i[tid] += s_i[tid + off];
            s_b[tid] += s_b[tid + off];
            s_m[tid] += s_m[tid + off];
        }
        __syncthreads();
    }
    if (tid == 0) {
        rowsum[row] = s_i[0];
        atomicAdd(acc + 1, s_b[0]);
        atomicAdd(acc + 2, s_m[0]);
    }
}

__global__ __launch_bounds__(256) void sim_kl_kernel(
    const signed char* __restrict__ c8, const int* __restrict__ rowsum,
    const float* __restrict__ Xi, const float* __restrict__ Xt,
    const float* __restrict__ Xc, float* __restrict__ acc)
{
    __shared__ float s_sim[16 * 512];
    __shared__ float s_redA[256];
    __shared__ float s_redB[256];
    __shared__ float s_stat1[16];
    __shared__ float s_stat2[16];
    __shared__ float s_tlogt[16];

    const int tid     = threadIdx.x;
    const int lane    = tid & 31;
    const int wave    = tid >> 5;
    const int laneLow = lane & 15;
    const int hi      = lane >> 4;
    const int m0      = blockIdx.x * 16;

    for (int tile = 0; tile < 4; ++tile) {
        const int n0 = (wave * 4 + tile) * 16;
        v8i accv = {};
        #pragma unroll
        for (int kc = 0; kc < 4; ++kc) {
            const int k0 = kc * 64;
            v8i a, b;
            #pragma unroll
            for (int v = 0; v < 8; ++v) {
                const int Ka = k0 + (v >> 1) * 16 + hi * 8 + (v & 1) * 4;
                a[v] = *(const int*)(c8 + (m0 + laneLow) * 256 + Ka);
                const int Kb = k0 + (v >> 1) * 16 + hi * 8 + (v & 1) * 4;
                b[v] = *(const int*)(c8 + (n0 + laneLow) * 256 + Kb);
            }
            accv = __builtin_amdgcn_wmma_i32_16x16x64_iu8(
                false, a, false, b, accv, false, false);
        }
        const int rsN = rowsum[n0 + laneLow];
        #pragma unroll
        for (int r = 0; r < 8; ++r) {
            const int row   = r + 8 * hi;
            const int inter = accv[r];
            const int uni   = rowsum[m0 + row] + rsN - inter;
            const float sim = (uni > 0) ? (float)inter / (float)uni : 0.0f;
            s_sim[row * 512 + n0 + laneLow] = sim * 5.0f;
        }
    }
    __syncthreads();

    const int r = tid >> 4;
    const int j = tid & 15;
    float* srow = s_sim + r * 512;

    float m = NEG_BIG;
    for (int c = j; c < 512; c += 16) m = fmaxf(m, srow[c]);
    s_redA[r * 16 + j] = m;
    __syncthreads();
    if (j == 0) {
        float mm = s_redA[r * 16];
        for (int q = 1; q < 16; ++q) mm = fmaxf(mm, s_redA[r * 16 + q]);
        s_stat1[r] = mm;
    }
    __syncthreads();
    const float rowmax = s_stat1[r];

    float se = 0.0f, ds = 0.0f;
    for (int c = j; c < 512; c += 16) {
        const float s = srow[c];
        const float e = expf(s - rowmax);
        se += e;
        ds += s * e;
    }
    s_redA[r * 16 + j] = se;
    s_redB[r * 16 + j] = ds;
    __syncthreads();
    if (j == 0) {
        float S = 0.0f, D = 0.0f;
        for (int q = 0; q < 16; ++q) { S += s_redA[r * 16 + q]; D += s_redB[r * 16 + q]; }
        s_stat2[r] = S;
        s_tlogt[r] = D / S - rowmax - logf(S);
    }
    __syncthreads();
    const float invS = 1.0f / s_stat2[r];
    for (int c = j; c < 512; c += 16)
        srow[c] = expf(srow[c] - rowmax) * invS;
    __syncthreads();

    const float wts[3] = {0.5f, 0.5f, 0.2f};
    const float* Xs[3] = {Xi, Xt, Xc};
    for (int mm = 0; mm < 3; ++mm) {
        const float* Xrow = Xs[mm] + (m0 + r) * 512;

        float xm = NEG_BIG;
        for (int c = j; c < 512; c += 16) xm = fmaxf(xm, Xrow[c]);
        s_redA[r * 16 + j] = xm;
        __syncthreads();
        if (j == 0) {
            float mx = s_redA[r * 16];
            for (int q = 1; q < 16; ++q) mx = fmaxf(mx, s_redA[r * 16 + q]);
            s_stat1[r] = mx;
        }
        __syncthreads();
        xm = s_stat1[r];

        float sex = 0.0f, dot = 0.0f;
        for (int c = j; c < 512; c += 16) {
            const float x = Xrow[c];
            sex += expf(x - xm);
            dot += srow[c] * x;
        }
        s_redA[r * 16 + j] = sex;
        s_redB[r * 16 + j] = dot;
        __syncthreads();
        if (j == 0) {
            float S2 = 0.0f, D2 = 0.0f;
            for (int q = 0; q < 16; ++q) { S2 += s_redA[r * 16 + q]; D2 += s_redB[r * 16 + q]; }
            const float term = s_tlogt[r] - D2 + xm + logf(S2);
            atomicAdd(acc, wts[mm] * term * (1.0f / 512.0f));
        }
        __syncthreads();
    }
}

__global__ void finalize_kernel(const float* acc, float* out) {
    if (threadIdx.x == 0)
        out[0] = acc[0] + 0.2f * acc[1] / (acc[2] + 1e-8f);
}

extern "C" void kernel_launch(void* const* d_in, const int* in_sizes, int n_in,
                              void* d_out, int out_size, void* d_ws, size_t ws_size,
                              hipStream_t stream) {
    const float* Xi   = (const float*)d_in[0];
    const float* Xt   = (const float*)d_in[1];
    const float* clog = (const float*)d_in[2];
    const float* Xc   = (const float*)d_in[3];
    const int*   mc   = (const int*)d_in[4];
    float* out = (float*)d_out;

    char* ws = (char*)d_ws;
    signed char* c8    = (signed char*)ws;
    int*         rsum  = (int*)(ws + 131072);
    float*       acc   = (float*)(ws + 133120);

    zero_acc_kernel<<<1, 32, 0, stream>>>(acc);
    pack_bce_kernel<<<512, 256, 0, stream>>>(mc, clog, c8, rsum, acc);
    sim_kl_kernel<<<32, 256, 0, stream>>>(c8, rsum, Xi, Xt, Xc, acc);
    finalize_kernel<<<1, 32, 0, stream>>>(acc, out);
}
